// TokenRouter_55379308315178
// MI455X (gfx1250) — matrix operand order repaired and checked by reading, not compiled, not run on hardware
//
#include <hip/hip_runtime.h>
#include <math.h>

// ---------------- CDNA5 (gfx1250) wave32 WMMA token-router ----------------
typedef __attribute__((ext_vector_type(16))) __bf16        v16bf;
typedef __attribute__((ext_vector_type(8)))  float         v8f;
typedef __attribute__((ext_vector_type(4)))  float         f32x4;
typedef __attribute__((ext_vector_type(4)))  unsigned int  u32x4;

#define D_DIM 1024            // model dim
#define H_DIM 512             // router hidden
#define E_DIM 8               // experts
#define KT    (D_DIM / 32)    // 32 K-tiles of 32 (bf16 WMMA K)
#define NT    (H_DIM / 16)    // 32 N-tiles of 16
#define TOK_PER_BLK 16
#define XROW  2080            // bytes per LDS xn row (1024 bf16 + 32B pad)
#define HROW  520             // floats per LDS h row (512 + 8 pad)

__device__ __forceinline__ unsigned f2bf(float f) {
  unsigned u = __builtin_bit_cast(unsigned, f);
  return (u + 0x7FFFu + ((u >> 16) & 1u)) >> 16;   // round-to-nearest-even bf16
}

// Pack W1 (D x H, f32 row-major) into per-lane WMMA B-fragment layout (bf16):
// block (kt,nt) = 32 lanes x 16 halves; lane<16 -> N=nt*16+lane, K=kt*32+i
//                                       lane>=16 -> same N, K=kt*32+16+i
__global__ void pack_w1_kernel(const float* __restrict__ W1,
                               unsigned short* __restrict__ out) {
  int idx  = blockIdx.x * blockDim.x + threadIdx.x;  // (kt*NT+nt)*32+lane
  int lane = idx & 31;
  int tile = idx >> 5;
  int nt   = tile & (NT - 1);
  int kt   = tile >> 5;
  if (kt >= KT) return;
  int n     = nt * 16 + (lane & 15);
  int kbase = kt * 32 + ((lane >> 4) << 4);
  unsigned short* dst = out + (size_t)idx * 16;
#pragma unroll
  for (int i = 0; i < 16; ++i)
    dst[i] = (unsigned short)f2bf(W1[(size_t)(kbase + i) * H_DIM + n]);
}

__global__ __launch_bounds__(128, 2)
void token_router_kernel(const float* __restrict__ x,
                         const float* __restrict__ gamma,
                         const float* __restrict__ beta,
                         const unsigned short* __restrict__ w1p,
                         const float* __restrict__ b1,
                         const float* __restrict__ W2,
                         const float* __restrict__ b2,
                         float* __restrict__ probsOut,
                         float* __restrict__ maskedOut) {
  __shared__ __align__(16) unsigned char ldsXn[TOK_PER_BLK * XROW];   // 33280 B
  __shared__ __align__(16) float         ldsH [TOK_PER_BLK * HROW];   // 33280 B
  __shared__ __align__(16) float         ldsW2[H_DIM * E_DIM];        // 16384 B
  __shared__ float                       ldsLogit[TOK_PER_BLK * E_DIM];

  const int  tid  = threadIdx.x;
  const int  lane = tid & 31;
  const int  wave = tid >> 5;                  // 4 waves of 32 (wave32)
  const long tok0 = (long)blockIdx.x * TOK_PER_BLK;

  // Stage W2 into LDS (4096 floats).
  {
    const f32x4* src = (const f32x4*)W2;
    f32x4*       dst = (f32x4*)ldsW2;
    for (int i = tid; i < (H_DIM * E_DIM) / 4; i += 128) dst[i] = src[i];
  }

  // ---------- Phase 1: LayerNorm (f32) -> bf16 rows in LDS ----------
  f32x4 gv[8], bv[8];
#pragma unroll
  for (int q = 0; q < 8; ++q) {
    gv[q] = *(const f32x4*)(gamma + lane * 32 + q * 4);
    bv[q] = *(const f32x4*)(beta  + lane * 32 + q * 4);
  }
#pragma unroll
  for (int i = 0; i < 4; ++i) {
    const int t = wave * 4 + i;
    const float* xr = x + (tok0 + t) * D_DIM + lane * 32;
    f32x4 xv[8];
    float s = 0.f, s2 = 0.f;
#pragma unroll
    for (int q = 0; q < 8; ++q) {
      xv[q] = *(const f32x4*)(xr + q * 4);
#pragma unroll
      for (int c = 0; c < 4; ++c) { float v = xv[q][c]; s += v; s2 += v * v; }
    }
#pragma unroll
    for (int m = 16; m >= 1; m >>= 1) {        // wave32 butterfly reduction
      s  += __shfl_xor(s,  m, 32);
      s2 += __shfl_xor(s2, m, 32);
    }
    const float mean = s  * (1.f / D_DIM);
    const float var  = s2 * (1.f / D_DIM) - mean * mean;
    const float rstd = rsqrtf(var + 1e-5f);
    unsigned w[16];
#pragma unroll
    for (int q = 0; q < 8; ++q) {
#pragma unroll
      for (int c = 0; c < 4; c += 2) {
        float v0 = (xv[q][c]     - mean) * rstd * gv[q][c]     + bv[q][c];
        float v1 = (xv[q][c + 1] - mean) * rstd * gv[q][c + 1] + bv[q][c + 1];
        w[q * 2 + (c >> 1)] = f2bf(v0) | (f2bf(v1) << 16);
      }
    }
    u32x4* dst = (u32x4*)(ldsXn + t * XROW + lane * 64);
#pragma unroll
    for (int q = 0; q < 4; ++q) {
      u32x4 pk = {w[q * 4 + 0], w[q * 4 + 1], w[q * 4 + 2], w[q * 4 + 3]};
      dst[q] = pk;
    }
  }
  __syncthreads();

  // ---------- Phase 2: GEMM1 via v_wmma_f32_16x16x32_bf16 ----------
  // Wave `wave` owns N-tiles [wave*8, wave*8+8); C = 8 x v8f fp32 accumulators.
  union AB { v16bf bf; f32x4 q[2]; };
  v8f acc[8] = {};
  // A-fragment (contiguous K, matching packed B): lane<16 -> M=lane, K {0..15}
  //                                    lane>=16 -> M=lane-16, K {16..31}
  const unsigned char* arow = ldsXn + (lane & 15) * XROW + ((lane >> 4) << 5);
  for (int kt = 0; kt < KT; ++kt) {
    AB a;
    a.q[0] = *(const f32x4*)(arow + kt * 64);
    a.q[1] = *(const f32x4*)(arow + kt * 64 + 16);
    if (kt + 1 < KT)   // hint next K-panel of W1 into cache (global_prefetch_b8)
      __builtin_prefetch(w1p + ((size_t)((kt + 1) * NT + wave * 8) * 32) * 16, 0, 0);
#pragma unroll
    for (int j = 0; j < 8; ++j) {
      AB b;
      const f32x4* bp =
          (const f32x4*)(w1p + ((size_t)(kt * NT + wave * 8 + j) * 32 + lane) * 16);
      b.q[0] = bp[0];
      b.q[1] = bp[1];
      acc[j] = __builtin_amdgcn_wmma_f32_16x16x32_bf16(
          /*neg_a=*/false, a.bf, /*neg_b=*/false, b.bf,
          /*c_mod=*/(short)0, acc[j], /*reuse_a=*/false, /*reuse_b=*/false);
    }
  }

  // Epilogue: +b1, exact GELU (erf), h -> LDS.  C layout: VGPR r -> M = r + (lane>=16)*8.
  const int rowHi = (lane >> 4) * 8;
  const int ncol  = lane & 15;
#pragma unroll
  for (int j = 0; j < 8; ++j) {
    const int   n    = (wave * 8 + j) * 16 + ncol;
    const float bias = b1[n];
#pragma unroll
    for (int r = 0; r < 8; ++r) {
      float v = acc[j][r] + bias;
      v = 0.5f * v * (1.0f + erff(v * 0.70710678118654752f));
      ldsH[(rowHi + r) * HROW + n] = v;
    }
  }
  __syncthreads();

  // ---------- Phase 3: GEMM2 (512 -> 8), one thread per (token, expert) ----------
  {
    const int t = tid >> 3;
    const int e = tid & 7;
    float a2 = b2[e];
    const float* hrow = ldsH + t * HROW;
#pragma unroll 8
    for (int hh = 0; hh < H_DIM; ++hh) a2 += hrow[hh] * ldsW2[hh * E_DIM + e];
    ldsLogit[t * E_DIM + e] = a2;           // TEMPERATURE == 1
  }
  __syncthreads();

  // ---------- Phase 4: top-2 mask + softmax, one lane per token ----------
  if (tid < TOK_PER_BLK) {
    float l[E_DIM];
#pragma unroll
    for (int e = 0; e < E_DIM; ++e) l[e] = ldsLogit[tid * E_DIM + e];
    int i1 = 0; float m1 = l[0];
#pragma unroll
    for (int e = 1; e < E_DIM; ++e) if (l[e] > m1) { m1 = l[e]; i1 = e; }
    int i2 = -1; float m2 = -INFINITY;
#pragma unroll
    for (int e = 0; e < E_DIM; ++e)
      if (e != i1 && l[e] > m2) { m2 = l[e]; i2 = e; }
    const float p2  = __expf(m2 - m1);
    const float inv = 1.f / (1.f + p2);
    const long  gt  = tok0 + tid;
#pragma unroll
    for (int e = 0; e < E_DIM; ++e) {
      const bool keep = (e == i1) || (e == i2);
      maskedOut[gt * E_DIM + e] = keep ? l[e] : -INFINITY;
      probsOut [gt * E_DIM + e] = (e == i1) ? inv : ((e == i2) ? p2 * inv : 0.f);
    }
  }
}

extern "C" void kernel_launch(void* const* d_in, const int* in_sizes, int n_in,
                              void* d_out, int out_size, void* d_ws, size_t ws_size,
                              hipStream_t stream) {
  const float* x     = (const float*)d_in[0];
  const float* gamma = (const float*)d_in[1];
  const float* beta  = (const float*)d_in[2];
  const float* W1    = (const float*)d_in[3];
  const float* b1    = (const float*)d_in[4];
  const float* W2    = (const float*)d_in[5];
  const float* b2    = (const float*)d_in[6];

  const long tokens = (long)in_sizes[0] / D_DIM;      // B*S = 32768
  unsigned short* w1p = (unsigned short*)d_ws;        // 1 MiB packed bf16 W1

  const int packThreads = KT * NT * 32;               // 32768
  pack_w1_kernel<<<packThreads / 256, 256, 0, stream>>>(W1, w1p);

  float* probsOut  = (float*)d_out;
  float* maskedOut = (float*)d_out + tokens * E_DIM;
  token_router_kernel<<<dim3((unsigned)(tokens / TOK_PER_BLK)), dim3(128), 0, stream>>>(
      x, gamma, beta, w1p, b1, W2, b2, probsOut, maskedOut);
}
